// Attention_7464653160540
// MI455X (gfx1250) — matrix operand order repaired and checked by reading, not compiled, not run on hardware
//
#include <hip/hip_runtime.h>
#include <math.h>

typedef __attribute__((ext_vector_type(16))) __bf16 v16bf;
typedef __attribute__((ext_vector_type(8)))  float  v8f;

#define EPS_F 1.1920928955078125e-07f

// ---------------------------------------------------------------------------
// CDNA5 async global->LDS copy (16B per lane), tracked by ASYNCcnt.
// ---------------------------------------------------------------------------
__device__ __forceinline__ void async_copy_b128(const void* gptr, void* lptr) {
  unsigned lds_off = (unsigned)(size_t)lptr;  // low 32 bits of shared ptr = LDS offset
  asm volatile("global_load_async_to_lds_b128 %0, %1, off"
               :: "v"(lds_off), "v"(gptr) : "memory");
}
__device__ __forceinline__ void wait_async0() {
  asm volatile("s_wait_asynccnt 0x0" ::: "memory");
}

// ---------------------------------------------------------------------------
// RMSNorm: per row (1024 elems), write bf16
// ---------------------------------------------------------------------------
__global__ __launch_bounds__(256) void rmsnorm_kernel(
    const float* __restrict__ x, const float* __restrict__ w,
    __bf16* __restrict__ out, int D) {
  const int row = blockIdx.x;
  const int tid = threadIdx.x, lane = tid & 31, wid = tid >> 5;
  const float* xr = x + (size_t)row * D;

  float ss = 0.f;
  for (int c = tid; c < D; c += 256) { float v = xr[c]; ss += v * v; }
  #pragma unroll
  for (int m = 16; m >= 1; m >>= 1) ss += __shfl_xor(ss, m, 32);

  __shared__ float red[8];
  if (lane == 0) red[wid] = ss;
  __syncthreads();
  if (wid == 0) {
    float t = (lane < 8) ? red[lane] : 0.f;
    #pragma unroll
    for (int m = 4; m >= 1; m >>= 1) t += __shfl_xor(t, m, 32);
    if (lane == 0) red[0] = t;
  }
  __syncthreads();
  const float scale = rsqrtf(red[0] / (float)D + EPS_F);
  for (int c = tid; c < D; c += 256)
    out[(size_t)row * D + c] = (__bf16)(xr[c] * scale * w[c]);
}

// ---------------------------------------------------------------------------
// Transpose + fp32->bf16: dst[n*K+k] = src[k*N+n]  (weights, one-time)
// ---------------------------------------------------------------------------
__global__ __launch_bounds__(256) void transpose_cvt_kernel(
    const float* __restrict__ src, __bf16* __restrict__ dst, int K, int N) {
  int i = blockIdx.x * 256 + threadIdx.x;
  if (i < N * K) {
    int n = i / K, k = i - n * K;
    dst[i] = (__bf16)src[(size_t)k * N + n];
  }
}

// ---------------------------------------------------------------------------
// WMMA fragment helpers (bf16 16x16x32, wave32 layouts, ISA 7.12.2)
// A 16x32 (row-major in LDS): elem e: j=e>>1,t=e&1 -> k=(j>=4?16:0)+hi*8+(j&3)*2+t
// B 32x16 read from K-contiguous (transposed) storage: elem e -> k = (e>=8?16:0)+hi*8+(e&7)
// C/D 16x16: vgpr r: row = r + hi*8, col = lane&15
// ---------------------------------------------------------------------------
__device__ __forceinline__ v16bf load_a_frag(const __bf16* p, int stride,
                                             int m15, int hi) {
  v16bf a;
  #pragma unroll
  for (int e = 0; e < 16; ++e) {
    int j = e >> 1, t = e & 1;
    int k = ((j & 4) << 2) + hi * 8 + ((j & 3) << 1) + t;
    a[e] = p[m15 * stride + k];
  }
  return a;
}
// p points at row n15=0 of an [n][k] tile; two 8-elem runs per lane half (A order).
__device__ __forceinline__ v16bf load_bt_frag(const __bf16* p, int stride,
                                              int n15, int hi) {
  v16bf b;
  #pragma unroll
  for (int e = 0; e < 16; ++e) b[e] = p[n15 * stride + ((e & 8) << 1) + hi * 8 + (e & 7)];
  return b;
}

// ---------------------------------------------------------------------------
// Tiled bf16 WMMA GEMM: C[M,N] = A[M,K] * Bt[N,K]^T (+ bias), out bf16 or f32
// Block tile 128x128x32; double-buffered LDS filled by async-to-LDS copies.
// 8 waves (4x2); wave tile 32x64 = 2x4 wmma tiles -> 8 wmma per k-step.
// ---------------------------------------------------------------------------
template <bool BIAS, bool OUT_BF16>
__global__ __launch_bounds__(256) void gemm_wmma_kernel(
    const __bf16* __restrict__ A, const __bf16* __restrict__ Bt,
    float* __restrict__ Cf, __bf16* __restrict__ Cb,
    const float* __restrict__ bias, int M, int N, int K) {
  constexpr int BM = 128, BN = 128, BK = 32;
  constexpr int LD = BK + 8;  // 40 elems = 80B rows (16B aligned)
  __shared__ __bf16 As[2][BM * LD];
  __shared__ __bf16 Bs[2][BN * LD];

  const int tid = threadIdx.x, lane = tid & 31, wid = tid >> 5;
  const int wm = wid >> 1, wn = wid & 1;     // 4 x 2 waves
  const int m15 = lane & 15, hi = lane >> 4;
  const int bm0 = blockIdx.y * BM;
  const int bn0 = blockIdx.x * BN;

  v8f acc[2][4];
  #pragma unroll
  for (int i = 0; i < 2; ++i)
    #pragma unroll
    for (int j = 0; j < 4; ++j) acc[i][j] = (v8f){0.f,0.f,0.f,0.f,0.f,0.f,0.f,0.f};

  const int la = tid * 2;
  auto stage = [&](int buf, int k0) {
    #pragma unroll
    for (int i = 0; i < 2; ++i) {  // A tile: 128 rows x 32 k, 512 x 16B
      int row = (la + i) >> 2, seg = (la + i) & 3;
      async_copy_b128(A + (size_t)(bm0 + row) * K + k0 + seg * 8,
                      &As[buf][row * LD + seg * 8]);
    }
    #pragma unroll
    for (int i = 0; i < 2; ++i) {  // Bt tile: 128 n-rows x 32 k, 512 x 16B
      int row = (la + i) >> 2, seg = (la + i) & 3;
      async_copy_b128(Bt + (size_t)(bn0 + row) * K + k0 + seg * 8,
                      &Bs[buf][row * LD + seg * 8]);
    }
  };

  const int nk = K / BK;
  stage(0, 0);
  for (int kt = 0; kt < nk; ++kt) {
    const int buf = kt & 1;
    wait_async0();      // own async transfers landed in LDS
    __syncthreads();    // everyone's transfers published; prev reads done
    if (kt + 1 < nk) stage(buf ^ 1, (kt + 1) * BK);
    if (kt + 2 < nk) {  // L2 prefetch two tiles ahead
      int k2 = (kt + 2) * BK;
      int row = la >> 2;
      __builtin_prefetch(A + (size_t)(bm0 + row) * K + k2, 0, 3);
      __builtin_prefetch(Bt + (size_t)(bn0 + row) * K + k2, 0, 3);
    }

    v16bf af[2], bf4[4];
    #pragma unroll
    for (int tm = 0; tm < 2; ++tm)
      af[tm] = load_a_frag(&As[buf][(wm * 32 + tm * 16) * LD], LD, m15, hi);
    #pragma unroll
    for (int tn = 0; tn < 4; ++tn)
      bf4[tn] = load_bt_frag(&Bs[buf][(wn * 64 + tn * 16) * LD], LD, m15, hi);

    #pragma unroll
    for (int tm = 0; tm < 2; ++tm)
      #pragma unroll
      for (int tn = 0; tn < 4; ++tn)
        acc[tm][tn] = __builtin_amdgcn_wmma_f32_16x16x32_bf16(
            false, af[tm], false, bf4[tn], (short)0, acc[tm][tn], false, false);
  }

  #pragma unroll
  for (int tm = 0; tm < 2; ++tm)
    #pragma unroll
    for (int tn = 0; tn < 4; ++tn)
      #pragma unroll
      for (int r = 0; r < 8; ++r) {
        int row = bm0 + wm * 32 + tm * 16 + hi * 8 + r;
        int col = bn0 + wn * 64 + tn * 16 + m15;
        float v = acc[tm][tn][r];
        if (BIAS) v += bias[col];
        if (OUT_BF16) Cb[(size_t)row * N + col] = (__bf16)v;
        else          Cf[(size_t)row * N + col] = v;
      }
}

// ---------------------------------------------------------------------------
// Flash attention with fused RoPE. qkv: bf16 [B*N, 3072] (q|k|v per head).
// Block: 64 q-rows of one (b,h); 4 waves; 32-key blocks staged in LDS
// (K via RoPE compute path, V via async-to-LDS). QK^T and PV in wmma bf16.
// ---------------------------------------------------------------------------
__global__ __launch_bounds__(128) void attn_kernel(
    const __bf16* __restrict__ qkv, const float* __restrict__ sin_t,
    const float* __restrict__ cos_t, __bf16* __restrict__ out, int Nn) {
  const int qb = blockIdx.x;
  const int h  = blockIdx.y;
  const int b  = blockIdx.z;
  const int tid = threadIdx.x, lane = tid & 31, w = tid >> 5;
  const int m15 = lane & 15, hi = lane >> 4;

  constexpr int LDK = 64 + 8;  // 72 elems = 144B rows (16B aligned)
  constexpr int LDP = 32 + 8;
  __shared__ __bf16 Ks[32 * LDK];
  __shared__ __bf16 Vs[32 * LDK];
  __shared__ __bf16 Ps[4][16 * LDP];

  const size_t rs = 3072;
  const __bf16* qbase = qkv + (size_t)b * Nn * rs + h * 64;
  const __bf16* kbase = qbase + 1024;
  const __bf16* vbase = qbase + 2048;

  // Preload Q A-fragments (16 rows x 64 d) with RoPE applied on the fly.
  const int qrow = qb * 64 + w * 16 + m15;
  v16bf qa[2];
  #pragma unroll
  for (int ch = 0; ch < 2; ++ch) {
    #pragma unroll
    for (int e = 0; e < 16; ++e) {
      int j = e >> 1, t = e & 1;
      int d = ch * 32 + ((j & 4) << 2) + hi * 8 + ((j & 3) << 1) + t;
      float x1 = (float)qbase[(size_t)qrow * rs + d];
      float x2 = (float)qbase[(size_t)qrow * rs + ((d + 32) & 63)];
      float rot = (d < 32) ? -x2 : x2;
      qa[ch][e] = (__bf16)(x1 * cos_t[qrow * 64 + d] + rot * sin_t[qrow * 64 + d]);
    }
  }

  v8f o[4];
  #pragma unroll
  for (int t = 0; t < 4; ++t) o[t] = (v8f){0.f,0.f,0.f,0.f,0.f,0.f,0.f,0.f};
  float mrow[8], lrow[8];
  #pragma unroll
  for (int r = 0; r < 8; ++r) { mrow[r] = -__builtin_inff(); lrow[r] = 0.f; }

  const int nkb = 2 * qb + 2;
  for (int kb = 0; kb < nkb; ++kb) {
    __syncthreads();  // previous iteration's Ks/Vs reads complete
    {  // V block (32 keys x 64 d): async-to-LDS, 2 x 16B per thread
      int lin = tid * 2;
      #pragma unroll
      for (int i = 0; i < 2; ++i) {
        int key = (lin + i) >> 3, seg = (lin + i) & 7;
        async_copy_b128(vbase + (size_t)(kb * 32 + key) * rs + seg * 8,
                        &Vs[key * LDK + seg * 8]);
      }
    }
    {  // K block with RoPE (needs VALU): 16 elems per thread
      int key = tid >> 2;
      int d0 = (tid & 3) * 16;
      int keyg = kb * 32 + key;
      const __bf16* ksrc = kbase + (size_t)keyg * rs;
      #pragma unroll
      for (int e = 0; e < 16; ++e) {
        int d = d0 + e;
        float x1 = (float)ksrc[d];
        float x2 = (float)ksrc[(d + 32) & 63];
        float rot = (d < 32) ? -x2 : x2;
        Ks[key * LDK + d] =
            (__bf16)(x1 * cos_t[keyg * 64 + d] + rot * sin_t[keyg * 64 + d]);
      }
    }
    wait_async0();
    __syncthreads();

    // S = Q K^T : two 16x16 tiles (keys 0-15, 16-31); Ks is [key][d] so the
    // B-operand (K^T) reads are d-contiguous -> b128 LDS loads.
    v8f s0 = (v8f){0.f,0.f,0.f,0.f,0.f,0.f,0.f,0.f};
    v8f s1 = s0;
    #pragma unroll
    for (int ch = 0; ch < 2; ++ch) {
      v16bf bk0 = load_bt_frag(&Ks[ch * 32], LDK, m15, hi);
      v16bf bk1 = load_bt_frag(&Ks[16 * LDK + ch * 32], LDK, m15, hi);
      s0 = __builtin_amdgcn_wmma_f32_16x16x32_bf16(false, qa[ch], false, bk0,
                                                   (short)0, s0, false, false);
      s1 = __builtin_amdgcn_wmma_f32_16x16x32_bf16(false, qa[ch], false, bk1,
                                                   (short)0, s1, false, false);
    }

    // online softmax over this 32-key block
    const float sc = 0.125f;
    float p0[8], p1[8];
    #pragma unroll
    for (int r = 0; r < 8; ++r) {
      int qg = qb * 64 + w * 16 + hi * 8 + r;
      int k0g = kb * 32 + m15;
      float f0 = (k0g > qg)      ? -__builtin_inff() : s0[r] * sc;
      float f1 = (k0g + 16 > qg) ? -__builtin_inff() : s1[r] * sc;
      float rmax = fmaxf(f0, f1);
      #pragma unroll
      for (int m = 8; m >= 1; m >>= 1) rmax = fmaxf(rmax, __shfl_xor(rmax, m, 32));
      float mnew = fmaxf(mrow[r], rmax);
      float alpha = __expf(mrow[r] - mnew);
      p0[r] = __expf(f0 - mnew);
      p1[r] = __expf(f1 - mnew);
      float rsum = p0[r] + p1[r];
      #pragma unroll
      for (int m = 8; m >= 1; m >>= 1) rsum += __shfl_xor(rsum, m, 32);
      lrow[r] = lrow[r] * alpha + rsum;
      mrow[r] = mnew;
      #pragma unroll
      for (int t = 0; t < 4; ++t) o[t][r] *= alpha;
    }

    // P (16x32) -> per-wave LDS, then reload in A layout
    #pragma unroll
    for (int r = 0; r < 8; ++r) {
      Ps[w][(hi * 8 + r) * LDP + m15]      = (__bf16)p0[r];
      Ps[w][(hi * 8 + r) * LDP + 16 + m15] = (__bf16)p1[r];
    }
    asm volatile("s_wait_dscnt 0x0" ::: "memory");  // same-wave LDS RAW

    v16bf pa = load_a_frag(&Ps[w][0], LDP, m15, hi);

    // O += P * V : 4 tiles over Dh (B element (k=key, n=d) -> strided reads)
    #pragma unroll
    for (int t = 0; t < 4; ++t) {
      v16bf bv;
      #pragma unroll
      for (int e = 0; e < 16; ++e)
        bv[e] = Vs[(((e & 8) << 1) + hi * 8 + (e & 7)) * LDK + t * 16 + m15];
      o[t] = __builtin_amdgcn_wmma_f32_16x16x32_bf16(false, pa, false, bv,
                                                     (short)0, o[t], false, false);
    }
  }

  #pragma unroll
  for (int t = 0; t < 4; ++t)
    #pragma unroll
    for (int r = 0; r < 8; ++r) {
      int row = qb * 64 + w * 16 + hi * 8 + r;
      int d = t * 16 + m15;
      float v = o[t][r] / lrow[r];
      out[((size_t)b * Nn + row) * 1024 + h * 64 + d] = (__bf16)v;
    }
}

// ---------------------------------------------------------------------------
// Launch: rmsnorm -> transpose weights -> QKV GEMM -> attention -> out GEMM
// ---------------------------------------------------------------------------
extern "C" void kernel_launch(void* const* d_in, const int* in_sizes, int n_in,
                              void* d_out, int out_size, void* d_ws, size_t ws_size,
                              hipStream_t stream) {
  const float* x      = (const float*)d_in[0];
  const float* norm_w = (const float*)d_in[1];
  const float* w_qkv  = (const float*)d_in[2];  // [1024,3072]
  const float* w_o    = (const float*)d_in[3];  // [1024,1024]
  const float* b_o    = (const float*)d_in[4];
  const float* sin_t  = (const float*)d_in[5];
  const float* cos_t  = (const float*)d_in[6];

  const int Bsz = 2, Nn = 2048, D = 1024;
  const int M = Bsz * Nn;  // 4096

  __bf16* ws      = (__bf16*)d_ws;
  __bf16* xn      = ws;                        // [4096,1024]
  __bf16* wqkv_t  = xn + (size_t)M * D;        // [3072,1024] (transposed)
  __bf16* wo_t    = wqkv_t + (size_t)3072 * D; // [1024,1024] (transposed)
  __bf16* qkv     = wo_t + (size_t)D * D;      // [4096,3072]
  __bf16* attn_o  = qkv + (size_t)M * 3072;    // [4096,1024]

  rmsnorm_kernel<<<M, 256, 0, stream>>>(x, norm_w, xn, D);
  transpose_cvt_kernel<<<(D * 3072 + 255) / 256, 256, 0, stream>>>(w_qkv, wqkv_t,
                                                                   D, 3072);
  transpose_cvt_kernel<<<(D * D + 255) / 256, 256, 0, stream>>>(w_o, wo_t, D, D);

  gemm_wmma_kernel<false, true><<<dim3(3072 / 128, M / 128), 256, 0, stream>>>(
      xn, wqkv_t, nullptr, qkv, nullptr, M, 3072, D);

  attn_kernel<<<dim3(Nn / 64, 16, Bsz), 128, 0, stream>>>(qkv, sin_t, cos_t,
                                                          attn_o, Nn);

  gemm_wmma_kernel<true, false><<<dim3(D / 128, M / 128), 256, 0, stream>>>(
      attn_o, wo_t, (float*)d_out, nullptr, b_o, M, D, D);
}
